// MVGRL_18691697672631
// MI455X (gfx1250) — matrix operand order repaired and checked by reading, not compiled, not run on hardware
//
#include <hip/hip_runtime.h>
#include <hip/hip_bf16.h>

typedef __attribute__((ext_vector_type(16))) _Float16 v16h;
typedef __attribute__((ext_vector_type(8)))  float    v8f;

#define D 128

// ---------------------------------------------------------------------------
// Pack W (f32 row-major [k][n]) -> Wt (f16 row-major [n][k]) so that B
// fragments for v_wmma_f32_16x16x32_f16 are pairs of contiguous 16B runs in LDS.
// ---------------------------------------------------------------------------
__global__ void pack_wt_kernel(const float* __restrict__ W, _Float16* __restrict__ Wt) {
    int idx = blockIdx.x * blockDim.x + threadIdx.x;
    if (idx < D * D) {
        int n = idx / D, k = idx % D;
        Wt[idx] = (_Float16)W[k * D + n];
    }
}

// ---------------------------------------------------------------------------
// Weighted in-degree: deg[dst[e]] += w[e]
// ---------------------------------------------------------------------------
__global__ void deg_kernel(const int* __restrict__ dst, const float* __restrict__ w,
                           float* deg, int E) {
    int e = blockIdx.x * blockDim.x + threadIdx.x;
    if (e < E)
        __hip_atomic_fetch_add(&deg[dst[e]], w[e], __ATOMIC_RELAXED, __HIP_MEMORY_SCOPE_AGENT);
}

// deg -> dinv in place
__global__ void dinv_kernel(float* deg, int n) {
    int i = blockIdx.x * blockDim.x + threadIdx.x;
    if (i < n) {
        float d = deg[i];
        deg[i] = d > 0.0f ? rsqrtf(d) : 0.0f;
    }
}

__global__ void norm_kernel(const int* __restrict__ src, const int* __restrict__ dst,
                            const float* __restrict__ w, const float* __restrict__ dinv,
                            float* __restrict__ norm, int E) {
    int e = blockIdx.x * blockDim.x + threadIdx.x;
    if (e < E) norm[e] = dinv[src[e]] * w[e] * dinv[dst[e]];
}

// ---------------------------------------------------------------------------
// WMMA GEMM: Y[r][n] = sum_k Xin[gather(r)][k] * W[k][n] + bias[n]
// Wt is W transposed in f16 ([n][k]).  perm==nullptr -> identity gather.
// Block: 128 threads = 4 waves; each wave computes a 16x128 strip.
// ---------------------------------------------------------------------------
__global__ __launch_bounds__(128)
void gemm_wmma_kernel(const float* __restrict__ Xin, const int* __restrict__ perm,
                      const _Float16* __restrict__ Wt, const float* __restrict__ bias,
                      float* __restrict__ Yout, int nrows) {
    __shared__ _Float16 lw[D * D];            // 32 KB: W^T staged once per block
    {
        const uint4* s = (const uint4*)Wt;
        uint4*       d = (uint4*)lw;
        for (int i = threadIdx.x; i < (D * D) / 8; i += blockDim.x) d[i] = s[i];
    }
    __syncthreads();

    const int wave  = threadIdx.x >> 5;
    const int lane  = threadIdx.x & 31;
    const int mtile = blockIdx.x * 4 + wave;
    const int row0  = mtile * 16;
    if (row0 >= nrows) return;

    const int mr = lane & 15;                 // matrix row (A) / col (B,C)
    const int hi = lane >> 4;                 // lane group 0/1

    int arow = row0 + mr;
    if (arow >= nrows) arow = nrows - 1;      // N is a multiple of 16; safety only
    if (perm) arow = perm[arow];
    const float* abase = Xin + (long)arow * D + (hi ? 8 : 0);

    v8f acc[8];
#pragma unroll
    for (int nt = 0; nt < 8; ++nt) acc[nt] = (v8f){0, 0, 0, 0, 0, 0, 0, 0};

#pragma unroll
    for (int kt = 0; kt < 4; ++kt) {
        const float* ap = abase + kt * 32;
        // A fragment per ISA 16-bit 16x32 layout:
        // lanes 0-15: elems 0-7 = K 0-7,  elems 8-15 = K 16-23
        // lanes16-31: elems 0-7 = K 8-15, elems 8-15 = K 24-31
        const float4 f0 = *reinterpret_cast<const float4*>(ap);
        const float4 f1 = *reinterpret_cast<const float4*>(ap + 4);
        const float4 f2 = *reinterpret_cast<const float4*>(ap + 16);
        const float4 f3 = *reinterpret_cast<const float4*>(ap + 20);
        v16h a;
        a[0]  = (_Float16)f0.x; a[1]  = (_Float16)f0.y; a[2]  = (_Float16)f0.z; a[3]  = (_Float16)f0.w;
        a[4]  = (_Float16)f1.x; a[5]  = (_Float16)f1.y; a[6]  = (_Float16)f1.z; a[7]  = (_Float16)f1.w;
        a[8]  = (_Float16)f2.x; a[9]  = (_Float16)f2.y; a[10] = (_Float16)f2.z; a[11] = (_Float16)f2.w;
        a[12] = (_Float16)f3.x; a[13] = (_Float16)f3.y; a[14] = (_Float16)f3.z; a[15] = (_Float16)f3.w;

        const int koff = kt * 32 + (hi ? 8 : 0);
#pragma unroll
        for (int nt = 0; nt < 8; ++nt) {
            const int n = nt * 16 + mr;
            // B fragment: lanes0-15 K 0-7,16-23; lanes16-31 K 8-15,24-31 -> two 16B runs
            typedef __attribute__((ext_vector_type(8))) _Float16 v8h_t;
            const v8h_t b0 = *(const v8h_t*)&lw[n * D + koff];
            const v8h_t b1 = *(const v8h_t*)&lw[n * D + koff + 16];
            v16h bv = __builtin_shufflevector(b0, b1, 0, 1, 2, 3, 4, 5, 6, 7, 8, 9, 10, 11, 12, 13, 14, 15);
            acc[nt] = __builtin_amdgcn_wmma_f32_16x16x32_f16(
                false, a, false, bv, (short)0, acc[nt], false, false);
        }
    }

    // C/D layout: VGPR j -> M = j + 8*hi, N = mr
#pragma unroll
    for (int nt = 0; nt < 8; ++nt) {
        const int n  = nt * 16 + mr;
        const float b = bias[n];
#pragma unroll
        for (int j = 0; j < 8; ++j) {
            const int r = row0 + j + hi * 8;
            Yout[(long)r * D + n] = acc[nt][j] + b;
        }
    }
}

// ---------------------------------------------------------------------------
// Edge scatter: accum[dst[e]][:] += h[src[e]][:] * norm[e].  One wave/edge.
// ---------------------------------------------------------------------------
__global__ __launch_bounds__(256)
void scatter_kernel(const int* __restrict__ src, const int* __restrict__ dst,
                    const float* __restrict__ norm, const float* __restrict__ h,
                    float* accum, int E) {
    const int wv   = (blockIdx.x * blockDim.x + threadIdx.x) >> 5;
    const int lane = threadIdx.x & 31;
    if (wv >= E) return;
    if (wv + 2048 < E) __builtin_prefetch(src + wv + 2048, 0, 0);
    const int   s  = src[wv];
    const int   d  = dst[wv];
    const float nm = norm[wv];
    const float4 v = *reinterpret_cast<const float4*>(h + (long)s * D + lane * 4);
    float* p = accum + (long)d * D + lane * 4;
    __hip_atomic_fetch_add(p + 0, v.x * nm, __ATOMIC_RELAXED, __HIP_MEMORY_SCOPE_AGENT);
    __hip_atomic_fetch_add(p + 1, v.y * nm, __ATOMIC_RELAXED, __HIP_MEMORY_SCOPE_AGENT);
    __hip_atomic_fetch_add(p + 2, v.z * nm, __ATOMIC_RELAXED, __HIP_MEMORY_SCOPE_AGENT);
    __hip_atomic_fetch_add(p + 3, v.w * nm, __ATOMIC_RELAXED, __HIP_MEMORY_SCOPE_AGENT);
}

// ---------------------------------------------------------------------------
// PReLU: out[i] = in[i] >= 0 ? in[i] : alpha[i%128]*in[i]
// ---------------------------------------------------------------------------
__global__ void prelu_kernel(const float* __restrict__ in, const float* __restrict__ alpha,
                             float* __restrict__ out, long n) {
    long i = (long)blockIdx.x * blockDim.x + threadIdx.x;
    if (i < n) {
        float v = in[i];
        float a = alpha[(int)(i & (D - 1))];
        out[i] = v >= 0.0f ? v : a * v;
    }
}

// ---------------------------------------------------------------------------
// Column partial sums -> colsum (atomics); 256 rows per block.
// ---------------------------------------------------------------------------
__global__ __launch_bounds__(256)
void colsum_kernel(const float* __restrict__ z, float* colsum, int nrows) {
    const int col   = threadIdx.x & (D - 1);
    const int sub   = threadIdx.x >> 7;   // 0/1
    const int rbase = blockIdx.x * 256;
    int rend = rbase + 256;
    if (rend > nrows) rend = nrows;
    float p = 0.0f;
    for (int r = rbase + sub; r < rend; r += 2) p += z[(long)r * D + col];
    __hip_atomic_fetch_add(&colsum[col], p, __ATOMIC_RELAXED, __HIP_MEMORY_SCOPE_AGENT);
}

// g[n] = sum_k sigmoid(colsum[k]/N) * projW[k][n] + projb[n]  (one block, 128 threads)
__global__ void gproj_kernel(const float* __restrict__ colsum, const float* __restrict__ projW,
                             const float* __restrict__ projb, float* __restrict__ g, float invn) {
    const int n = threadIdx.x;
    float acc = projb[n];
    for (int k = 0; k < D; ++k) {
        float m = colsum[k] * invn;
        float s = 1.0f / (1.0f + __expf(-m));
        acc += s * projW[k * D + n];
    }
    g[n] = acc;
}

// ---------------------------------------------------------------------------
extern "C" void kernel_launch(void* const* d_in, const int* in_sizes, int n_in,
                              void* d_out, int out_size, void* d_ws, size_t ws_size,
                              hipStream_t stream) {
    (void)n_in; (void)out_size; (void)ws_size;

    const float* x     = (const float*)d_in[0];
    const int*   ei    = (const int*)d_in[1];
    const float* w     = (const float*)d_in[2];
    const int*   perm1 = (const int*)d_in[3];
    const int*   perm2 = (const int*)d_in[4];
    const float* W1a = (const float*)d_in[5],  *b1a = (const float*)d_in[6];
    const float* W2a = (const float*)d_in[7],  *b2a = (const float*)d_in[8];
    const float* al1 = (const float*)d_in[9];
    const float* W1b = (const float*)d_in[10], *b1b = (const float*)d_in[11];
    const float* W2b = (const float*)d_in[12], *b2b = (const float*)d_in[13];
    const float* al2 = (const float*)d_in[14];
    const float* projW = (const float*)d_in[15], *projb = (const float*)d_in[16];

    const int N = in_sizes[0] / D;
    const int E = in_sizes[2];
    const int* srcp = ei;
    const int* dstp = ei + E;

    // ---- workspace carve-out (256B aligned) ----
    size_t off = 0;
    auto carve = [&](size_t bytes) -> void* {
        void* p = (char*)d_ws + off;
        off += (bytes + 255) & ~(size_t)255;
        return p;
    };
    _Float16* wt1a = (_Float16*)carve(D * D * sizeof(_Float16));
    _Float16* wt2a = (_Float16*)carve(D * D * sizeof(_Float16));
    _Float16* wt1b = (_Float16*)carve(D * D * sizeof(_Float16));
    _Float16* wt2b = (_Float16*)carve(D * D * sizeof(_Float16));
    float* dinv   = (float*)carve((size_t)N * sizeof(float));     // deg then dinv
    float* normv  = (float*)carve((size_t)E * sizeof(float));
    float* bufA   = (float*)carve((size_t)N * D * sizeof(float));
    float* bufB   = (float*)carve((size_t)N * D * sizeof(float));
    float* colsum = (float*)carve(D * sizeof(float));

    float* out = (float*)d_out;
    const size_t ND = (size_t)N * D;
    float* z1  = out;
    float* z2  = out + ND;
    float* g1  = out + 2 * ND;
    float* g2  = out + 2 * ND + D;
    float* z1n = out + 2 * ND + 2 * D;
    float* z2n = out + 3 * ND + 2 * D;

    // ---- one-time setup: weight packing + symmetric norm ----
    pack_wt_kernel<<<(D * D + 255) / 256, 256, 0, stream>>>(W1a, wt1a);
    pack_wt_kernel<<<(D * D + 255) / 256, 256, 0, stream>>>(W2a, wt2a);
    pack_wt_kernel<<<(D * D + 255) / 256, 256, 0, stream>>>(W1b, wt1b);
    pack_wt_kernel<<<(D * D + 255) / 256, 256, 0, stream>>>(W2b, wt2b);

    hipMemsetAsync(dinv, 0, (size_t)N * sizeof(float), stream);
    deg_kernel<<<(E + 255) / 256, 256, 0, stream>>>(dstp, w, dinv, E);
    dinv_kernel<<<(N + 255) / 256, 256, 0, stream>>>(dinv, N);
    norm_kernel<<<(E + 255) / 256, 256, 0, stream>>>(srcp, dstp, w, dinv, normv, E);

    const int mtiles    = (N + 15) / 16;
    const int gemmGrid  = (mtiles + 3) / 4;
    const int scatGrid  = (E * 32 + 255) / 256;
    const int pGrid     = (int)((ND + 255) / 256);

    auto run_view = [&](const int* perm, const _Float16* wtA, const float* bA,
                        const _Float16* wtB, const float* bB,
                        const float* alpha, float* zout) {
        gemm_wmma_kernel<<<gemmGrid, 128, 0, stream>>>(x, perm, wtA, bA, bufA, N);
        hipMemsetAsync(bufB, 0, ND * sizeof(float), stream);
        scatter_kernel<<<scatGrid, 256, 0, stream>>>(srcp, dstp, normv, bufA, bufB, E);
        prelu_kernel<<<pGrid, 256, 0, stream>>>(bufB, alpha, bufB, (long)ND);
        gemm_wmma_kernel<<<gemmGrid, 128, 0, stream>>>(bufB, nullptr, wtB, bB, bufA, N);
        hipMemsetAsync(bufB, 0, ND * sizeof(float), stream);
        scatter_kernel<<<scatGrid, 256, 0, stream>>>(srcp, dstp, normv, bufA, bufB, E);
        prelu_kernel<<<pGrid, 256, 0, stream>>>(bufB, alpha, zout, (long)ND);
    };

    run_view(nullptr, wt1a, b1a, wt2a, b2a, al1, z1);   // z1
    run_view(nullptr, wt1b, b1b, wt2b, b2b, al2, z2);   // z2
    run_view(perm1,   wt1a, b1a, wt2a, b2a, al1, z1n);  // corrupted view 1
    run_view(perm2,   wt1b, b1b, wt2b, b2b, al2, z2n);  // corrupted view 2

    // ---- graph-level projections ----
    hipMemsetAsync(colsum, 0, D * sizeof(float), stream);
    colsum_kernel<<<(N + 255) / 256, 256, 0, stream>>>(z1, colsum, N);
    gproj_kernel<<<1, D, 0, stream>>>(colsum, projW, projb, g1, 1.0f / (float)N);

    hipMemsetAsync(colsum, 0, D * sizeof(float), stream);
    colsum_kernel<<<(N + 255) / 256, 256, 0, stream>>>(z2, colsum, N);
    gproj_kernel<<<1, D, 0, stream>>>(colsum, projW, projb, g2, 1.0f / (float)N);
}
